// LlamaAttention_40776419508478
// MI455X (gfx1250) — matrix operand order repaired and checked by reading, not compiled, not run on hardware
//
#include <hip/hip_runtime.h>

// ---------------- problem constants ----------------
constexpr int Bc = 2, Tc = 1024, Dc = 4096;
constexpr int Nc = 32, Kc = 8, Hc = 128;
constexpr int SLOTSc = 4096;
// -ln(500000)/64
#define NEG_LNTHETA_OVER64 (-0.20503692830701095f)
#define QSCALE 0.08838834764831845f   // 128^-0.5

// ---------------- CDNA5 WMMA / TDM types ----------------
typedef __attribute__((ext_vector_type(16))) __bf16 v16bf;
typedef __attribute__((ext_vector_type(8)))  float  v8f;
typedef __attribute__((ext_vector_type(4)))  unsigned int u32x4;
typedef __attribute__((ext_vector_type(8)))  int          i32x8;
typedef __attribute__((ext_vector_type(4)))  int          i32x4;
typedef unsigned int  u32;
typedef unsigned short u16;

union AFrag { v16bf v; u32 u[8]; };
union CAcc  { v8f   v; float f[8]; };

static __device__ inline v8f wmma_bf16(const AFrag& a, const AFrag& b, v8f c) {
  // D = A(16x32 bf16) * B(32x16 bf16) + C(16x16 f32)
  return __builtin_amdgcn_wmma_f32_16x16x32_bf16(false, a.v, false, b.v,
                                                 (short)0, c, false, false);
}

// native hardware f32 -> bf16 conversion
static __device__ inline u16 f2bf_bits(float f) {
  union { __bf16 b; u16 s; } u;
  u.b = (__bf16)f;
  return u.s;
}
// packed pair (lets the compiler form packed cvt + single b32 LDS store)
static __device__ inline u32 f2bf2(float lo, float hi) {
  union { __bf16 b[2]; u32 u; } u;
  u.b[0] = (__bf16)lo; u.b[1] = (__bf16)hi;
  return u.u;
}

// A-matrix (16x32 bf16) element-pair base K for vector slot v, lane group kg
static __device__ inline int a_k0(int v, int kg) {
  return 2 * v + 16 * kg;
}

// =====================================================================
// Tensor Data Mover: async 2D bf16 tile load Global -> LDS.
// rows x row_elems tile, row stride gstride_elems in global; LDS rows are
// padded 64 data DWORDs + 2 pad DWORDs (264B) to match u16[...][132] tiles.
// Descriptor fields per CDNA5 ISA.
// =====================================================================
static __device__ inline void tdm_load_2d_bf16(const u16* gsrc, u32 lds_byte,
                                               int rows, int row_elems,
                                               int gstride_elems) {
  unsigned long long ga = (unsigned long long)(uintptr_t)gsrc;
  u32x4 g0;
  g0.x = 1u;                                    // count=1 (valid), no gather
  g0.y = lds_byte;                              // lds_addr
  g0.z = (u32)(ga & 0xFFFFFFFFu);               // global_addr[31:0]
  g0.w = (u32)((ga >> 32) & 0x01FFFFFFu)        // global_addr[56:32]
         | (2u << 30);                          // type=2 ("image")
  i32x8 g1;
  g1[0] = (1 << 16)        // data_size = 2 bytes
        | (1 << 20)        // pad_enable
        | (5 << 22)        // pad_interval: 64 DWORDs
        | (1 << 25);       // pad_amount: 2 DWORDs
  g1[1] = (row_elems & 0xFFFF) << 16;                       // tensor_dim0[15:0]
  g1[2] = ((row_elems >> 16) & 0xFFFF) | ((rows & 0xFFFF) << 16);  // dim0 hi | dim1 lo
  g1[3] = ((rows >> 16) & 0xFFFF) | ((row_elems & 0xFFFF) << 16);  // dim1 hi | tile_dim0
  g1[4] = (rows & 0xFFFF);                                  // tile_dim1 (tile_dim2=0)
  g1[5] = gstride_elems;                                    // tensor_dim0_stride lo
  g1[6] = 0;
  g1[7] = 0;
  i32x4 z4 = {0, 0, 0, 0};
#if defined(__clang_major__) && (__clang_major__ >= 23)
  i32x8 z8 = {0, 0, 0, 0, 0, 0, 0, 0};
  __builtin_amdgcn_tensor_load_to_lds(g0, g1, z4, z4, z8, 0);
#else
  __builtin_amdgcn_tensor_load_to_lds(g0, g1, z4, z4, 0);
#endif
}

// =====================================================================
// Kernel 1: fused QKV projection + RoPE + cache scatter + bf16 emit
// grid.x = (B*T)/64 row tiles, grid.y = N + 2K head jobs, 128 threads
// =====================================================================
__global__ __launch_bounds__(128) void qkv_rope_cache_kernel(
    const float* __restrict__ x, const int* __restrict__ positions,
    const float* __restrict__ wq, const float* __restrict__ wk,
    const float* __restrict__ wv, const int* __restrict__ widx,
    float* __restrict__ kc_out, float* __restrict__ vc_out,
    u16* __restrict__ qb, u16* __restrict__ kb, u16* __restrict__ vb) {
  __shared__ __align__(16) u16 As[64][36];    // tokens x Dchunk (bf16)
  __shared__ __align__(16) u16 Bst[128][36];  // H x Dchunk (bf16, transposed)

  const int tid = threadIdx.x;
  const int lane = tid & 31;
  const int wid = tid >> 5;
  const int rt = blockIdx.x * 64;
  const int job = blockIdx.y;

  const float* w;
  if (job < Nc)            w = wq + (size_t)job * Dc * Hc;
  else if (job < Nc + Kc)  w = wk + (size_t)(job - Nc) * Dc * Hc;
  else                     w = wv + (size_t)(job - Nc - Kc) * Dc * Hc;

  CAcc acc[8];
#pragma unroll
  for (int j = 0; j < 8; ++j)
#pragma unroll
    for (int e = 0; e < 8; ++e) acc[j].f[e] = 0.f;

  const int arow = tid >> 1, ahalf = tid & 1;
  const int bdr = tid >> 2, bq = tid & 3;
  const int m = lane & 15, kg = lane >> 4;

  for (int kd = 0; kd < Dc; kd += 32) {
    // ---- stage A: x[rt..rt+63][kd..kd+31], float4 loads, packed bf16 stores
    {
      const float4* xs4 =
          (const float4*)(x + (size_t)(rt + arow) * Dc + kd + ahalf * 16);
      __builtin_prefetch(xs4 + 8, 0, 0);
#pragma unroll
      for (int c4 = 0; c4 < 4; ++c4) {
        const float4 vv = xs4[c4];
        *(u32*)&As[arow][ahalf * 16 + c4 * 4]     = f2bf2(vv.x, vv.y);
        *(u32*)&As[arow][ahalf * 16 + c4 * 4 + 2] = f2bf2(vv.z, vv.w);
      }
    }
    // ---- stage B (transposed): w[kd..kd+31][0..127] -> Bst[h][dchunk]
    {
      const float4* wsrc4 =
          (const float4*)(w + (size_t)(kd + bdr) * Hc + bq * 32);
      __builtin_prefetch(wsrc4 + 8 * Hc, 0, 0);
#pragma unroll
      for (int c4 = 0; c4 < 8; ++c4) {
        const float4 vv = wsrc4[c4];
        Bst[bq * 32 + c4 * 4 + 0][bdr] = f2bf_bits(vv.x);
        Bst[bq * 32 + c4 * 4 + 1][bdr] = f2bf_bits(vv.y);
        Bst[bq * 32 + c4 * 4 + 2][bdr] = f2bf_bits(vv.z);
        Bst[bq * 32 + c4 * 4 + 3][bdr] = f2bf_bits(vv.w);
      }
    }
    __syncthreads();

    AFrag a;
#pragma unroll
    for (int v = 0; v < 8; ++v)
      a.u[v] = *(const u32*)&As[wid * 16 + m][a_k0(v, kg)];
#pragma unroll
    for (int j = 0; j < 8; ++j) {
      AFrag bf;
#pragma unroll
      for (int v = 0; v < 8; ++v)
        bf.u[v] = *(const u32*)&Bst[j * 16 + m][2 * v + 16 * kg];
      acc[j].v = wmma_bf16(a, bf, acc[j].v);
    }
    __syncthreads();
  }

  // ---------------- epilogue ----------------
  const int rbase = rt + wid * 16;
  if (job < Nc) {  // Q head: RoPE + scale -> bf16 qb
    const int n = job;
#pragma unroll
    for (int v = 0; v < 8; ++v) {
      const int rg = rbase + v + 8 * kg;
      const int b = rg >> 10, t = rg & (Tc - 1);
      const float pos = (float)positions[rg];
      const size_t base = ((size_t)(b * Nc + n) * Tc + t) * Hc;
#pragma unroll
      for (int j = 0; j < 4; ++j) {
        const int h = j * 16 + m;
        const float ang = pos * __expf((float)h * NEG_LNTHETA_OVER64);
        const float cs = __cosf(ang), sn = __sinf(ang);
        const float x1 = acc[j].f[v], x2 = acc[j + 4].f[v];
        qb[base + h]      = f2bf_bits((x1 * cs - x2 * sn) * QSCALE);
        qb[base + h + 64] = f2bf_bits((x2 * cs + x1 * sn) * QSCALE);
      }
    }
  } else if (job < Nc + Kc) {  // K head: RoPE -> f32 cache + bf16 kb
    const int n2 = job - Nc;
#pragma unroll
    for (int v = 0; v < 8; ++v) {
      const int rg = rbase + v + 8 * kg;
      const int b = rg >> 10, t = rg & (Tc - 1);
      const float pos = (float)positions[rg];
      const int slot = widx[rg];
      const size_t cbase = ((size_t)n2 * SLOTSc + slot) * Hc;
      const size_t bbase = ((size_t)(b * Kc + n2) * Tc + t) * Hc;
#pragma unroll
      for (int j = 0; j < 4; ++j) {
        const int h = j * 16 + m;
        const float ang = pos * __expf((float)h * NEG_LNTHETA_OVER64);
        const float cs = __cosf(ang), sn = __sinf(ang);
        const float x1 = acc[j].f[v], x2 = acc[j + 4].f[v];
        const float r1 = x1 * cs - x2 * sn, r2 = x2 * cs + x1 * sn;
        kc_out[cbase + h] = r1;  kc_out[cbase + h + 64] = r2;
        kb[bbase + h] = f2bf_bits(r1);  kb[bbase + h + 64] = f2bf_bits(r2);
      }
    }
  } else {  // V head: f32 cache + bf16 vb
    const int n2 = job - Nc - Kc;
#pragma unroll
    for (int v = 0; v < 8; ++v) {
      const int rg = rbase + v + 8 * kg;
      const int b = rg >> 10, t = rg & (Tc - 1);
      const int slot = widx[rg];
      const size_t cbase = ((size_t)n2 * SLOTSc + slot) * Hc;
      const size_t bbase = ((size_t)(b * Kc + n2) * Tc + t) * Hc;
#pragma unroll
      for (int j = 0; j < 8; ++j) {
        const int h = j * 16 + m;
        const float val = acc[j].f[v];
        vc_out[cbase + h] = val;
        vb[bbase + h] = f2bf_bits(val);
      }
    }
  }
}

// =====================================================================
// Kernel 2: causal flash attention; Q/K tiles DMA'd by the TDM, V staged
// transposed by the waves while the DMA is in flight.
// grid.x = T/64 q tiles, grid.y = B*N, 128 threads (4 waves)
// =====================================================================
__global__ __launch_bounds__(128) void flash_attn_kernel(
    const u16* __restrict__ qb, const u16* __restrict__ kb,
    const u16* __restrict__ vb, u16* __restrict__ ab) {
  __shared__ __align__(16) u16 Qs[64][132];    // q rows x H (TDM-filled)
  __shared__ __align__(16) u16 Ks[64][132];    // keys x H (TDM-filled)
  __shared__ __align__(16) u16 Vst[128][68];   // H x keys (transposed)
  __shared__ __align__(16) u16 Ps[4][16][68];  // per-wave P tile

  const int tid = threadIdx.x;
  const int lane = tid & 31;
  const int wid = tid >> 5;
  const int qt = blockIdx.x;
  const int b = blockIdx.y / Nc;
  const int n = blockIdx.y % Nc;
  const int kh = n >> 2;  // GQA: N/K = 4
  const int qbase = qt * 64;

  const u16* qsrc = qb + ((size_t)(b * Nc + n) * Tc + qbase) * Hc;
  const u16* ksrc0 = kb + ((size_t)(b * Kc + kh) * Tc) * Hc;
  const u16* vsrc0 = vb + ((size_t)(b * Kc + kh) * Tc) * Hc;

  // ---- async Q tile load (64 x 128 bf16) via Tensor Data Mover
  if (wid == 0) {
    tdm_load_2d_bf16(qsrc, (u32)(uintptr_t)&Qs[0][0], 64, Hc, Hc);
    __builtin_amdgcn_s_wait_tensorcnt(0);
  }
  __syncthreads();

  const int m = lane & 15, kg = lane >> 4;

  AFrag aq[4];
#pragma unroll
  for (int kk = 0; kk < 4; ++kk)
#pragma unroll
    for (int v = 0; v < 8; ++v)
      aq[kk].u[v] = *(const u32*)&Qs[wid * 16 + m][kk * 32 + a_k0(v, kg)];

  CAcc co[8];
#pragma unroll
  for (int j = 0; j < 8; ++j)
#pragma unroll
    for (int e = 0; e < 8; ++e) co[j].f[e] = 0.f;
  float mrow[8], lrow[8];
#pragma unroll
  for (int v = 0; v < 8; ++v) { mrow[v] = -1e30f; lrow[v] = 0.f; }

  const int nS = qt + 1;  // causal
  for (int s = 0; s < nS; ++s) {
    const int sbase = s * 64;
    // ---- async K tile DMA, overlapped with manual V transpose staging
    if (wid == 0)
      tdm_load_2d_bf16(ksrc0 + (size_t)sbase * Hc, (u32)(uintptr_t)&Ks[0][0],
                       64, Hc, Hc);
    {
      const int key = tid >> 1, hf = tid & 1;
      const u32* vs32 = (const u32*)(vsrc0 + (size_t)(sbase + key) * Hc + hf * 64);
#pragma unroll
      for (int i = 0; i < 32; ++i) {
        const u32 pr = vs32[i];
        Vst[hf * 64 + 2 * i][key]     = (u16)pr;
        Vst[hf * 64 + 2 * i + 1][key] = (u16)(pr >> 16);
      }
    }
    if (wid == 0) __builtin_amdgcn_s_wait_tensorcnt(0);
    __syncthreads();

    // ---- S = Q * K^T  (4 key tiles x 4 H-chunks)
    CAcc sacc[4];
#pragma unroll
    for (int j = 0; j < 4; ++j) {
#pragma unroll
      for (int e = 0; e < 8; ++e) sacc[j].f[e] = 0.f;
#pragma unroll
      for (int kk = 0; kk < 4; ++kk) {
        AFrag bk;
#pragma unroll
        for (int v = 0; v < 8; ++v)
          bk.u[v] = *(const u32*)&Ks[j * 16 + m][kk * 32 + 2 * v + 16 * kg];
        sacc[j].v = wmma_bf16(aq[kk], bk, sacc[j].v);
      }
    }

    // ---- causal mask, online softmax update, P -> bf16 LDS
#pragma unroll
    for (int v = 0; v < 8; ++v) {
      const int rowg = qbase + wid * 16 + v + 8 * kg;
#pragma unroll
      for (int j = 0; j < 4; ++j) {
        const int keyg = sbase + j * 16 + m;
        if (keyg > rowg) sacc[j].f[v] = -1e30f;
      }
      float rv = sacc[0].f[v];
#pragma unroll
      for (int j = 1; j < 4; ++j) rv = fmaxf(rv, sacc[j].f[v]);
#pragma unroll
      for (int off = 1; off < 16; off <<= 1) rv = fmaxf(rv, __shfl_xor(rv, off));
      const float mn = fmaxf(mrow[v], rv);
      const float alpha = __expf(mrow[v] - mn);
      mrow[v] = mn;
      float rs = 0.f;
#pragma unroll
      for (int j = 0; j < 4; ++j) {
        const float p = __expf(sacc[j].f[v] - mn);
        sacc[j].f[v] = p;
        rs += p;
      }
#pragma unroll
      for (int off = 1; off < 16; off <<= 1) rs += __shfl_xor(rs, off);
      lrow[v] = lrow[v] * alpha + rs;
#pragma unroll
      for (int j = 0; j < 8; ++j) co[j].f[v] *= alpha;
#pragma unroll
      for (int j = 0; j < 4; ++j)
        Ps[wid][v + 8 * kg][j * 16 + m] = f2bf_bits(sacc[j].f[v]);
    }
    __syncthreads();

    // ---- O += P * V  (2 key chunks x 8 H tiles)
#pragma unroll
    for (int kc = 0; kc < 2; ++kc) {
      AFrag ap;
#pragma unroll
      for (int v = 0; v < 8; ++v)
        ap.u[v] = *(const u32*)&Ps[wid][m][kc * 32 + a_k0(v, kg)];
#pragma unroll
      for (int jo = 0; jo < 8; ++jo) {
        AFrag bv;
#pragma unroll
        for (int v = 0; v < 8; ++v)
          bv.u[v] = *(const u32*)&Vst[jo * 16 + m][kc * 32 + 2 * v + 16 * kg];
        co[jo].v = wmma_bf16(ap, bv, co[jo].v);
      }
    }
    __syncthreads();  // before next tile's DMA / staging overwrites LDS
  }

  // ---- normalize, emit attn bf16
#pragma unroll
  for (int v = 0; v < 8; ++v) {
    const float invl = 1.0f / lrow[v];
    const int rowg = qbase + wid * 16 + v + 8 * kg;
    const size_t base = ((size_t)(b * Nc + n) * Tc + rowg) * Hc;
#pragma unroll
    for (int jo = 0; jo < 8; ++jo)
      ab[base + jo * 16 + m] = f2bf_bits(co[jo].f[v] * invl);
  }
}

// =====================================================================
// Kernel 3: output projection  o[b,t,d] = sum_{n,h} attn * wo
// grid.x = (B*T)/64, grid.y = D/128, 128 threads
// =====================================================================
__global__ __launch_bounds__(128) void oproj_kernel(
    const u16* __restrict__ ab, const float* __restrict__ wo,
    float* __restrict__ o_out) {
  __shared__ __align__(16) u16 As[64][36];
  __shared__ __align__(16) u16 Bst[128][36];

  const int tid = threadIdx.x;
  const int lane = tid & 31;
  const int wid = tid >> 5;
  const int rt = blockIdx.x * 64;
  const int dbase = blockIdx.y * 128;

  CAcc acc[8];
#pragma unroll
  for (int j = 0; j < 8; ++j)
#pragma unroll
    for (int e = 0; e < 8; ++e) acc[j].f[e] = 0.f;

  const int arow = tid >> 1, ahalf = tid & 1;
  const int bdr = tid >> 2, bq = tid & 3;
  const int m = lane & 15, kg = lane >> 4;

  for (int nhb = 0; nhb < Nc * Hc; nhb += 32) {
    // ---- stage A: raw bf16 copy from attn (16 h's stay in one head)
    {
      const int rg = rt + arow;
      const int b = rg >> 10, t = rg & (Tc - 1);
      const int nh0 = nhb + ahalf * 16;
      const int n0 = nh0 >> 7, h0 = nh0 & 127;
      const uint2* src =
          (const uint2*)(ab + ((size_t)(b * Nc + n0) * Tc + t) * Hc + h0);
      uint2* dst = (uint2*)&As[arow][ahalf * 16];
#pragma unroll
      for (int i = 0; i < 4; ++i) dst[i] = src[i];
    }
    // ---- stage B (transposed) from wo, float4 loads
    {
      const float4* wsrc4 =
          (const float4*)(wo + (size_t)(nhb + bdr) * Dc + dbase + bq * 32);
      __builtin_prefetch(wsrc4 + 8 * Dc, 0, 0);
#pragma unroll
      for (int c4 = 0; c4 < 8; ++c4) {
        const float4 vv = wsrc4[c4];
        Bst[bq * 32 + c4 * 4 + 0][bdr] = f2bf_bits(vv.x);
        Bst[bq * 32 + c4 * 4 + 1][bdr] = f2bf_bits(vv.y);
        Bst[bq * 32 + c4 * 4 + 2][bdr] = f2bf_bits(vv.z);
        Bst[bq * 32 + c4 * 4 + 3][bdr] = f2bf_bits(vv.w);
      }
    }
    __syncthreads();

    AFrag a;
#pragma unroll
    for (int v = 0; v < 8; ++v)
      a.u[v] = *(const u32*)&As[wid * 16 + m][a_k0(v, kg)];
#pragma unroll
    for (int j = 0; j < 8; ++j) {
      AFrag bf;
#pragma unroll
      for (int v = 0; v < 8; ++v)
        bf.u[v] = *(const u32*)&Bst[j * 16 + m][2 * v + 16 * kg];
      acc[j].v = wmma_bf16(a, bf, acc[j].v);
    }
    __syncthreads();
  }

#pragma unroll
  for (int v = 0; v < 8; ++v) {
    const int rg = rt + wid * 16 + v + 8 * kg;
#pragma unroll
    for (int j = 0; j < 8; ++j)
      o_out[(size_t)rg * Dc + dbase + j * 16 + m] = acc[j].f[v];
  }
}

// =====================================================================
extern "C" void kernel_launch(void* const* d_in, const int* in_sizes, int n_in,
                              void* d_out, int out_size, void* d_ws, size_t ws_size,
                              hipStream_t stream) {
  const float* x         = (const float*)d_in[0];
  const int*   positions = (const int*)d_in[1];
  const float* wq        = (const float*)d_in[2];
  const float* wk        = (const float*)d_in[3];
  const float* wv        = (const float*)d_in[4];
  const float* wo        = (const float*)d_in[5];
  const float* kc_in     = (const float*)d_in[6];
  const float* vc_in     = (const float*)d_in[7];
  const int*   widx      = (const int*)d_in[8];

  constexpr size_t cacheElems = (size_t)Kc * SLOTSc * Hc;  // 4,194,304
  float* out    = (float*)d_out;
  float* kc_out = out;
  float* vc_out = out + cacheElems;
  float* o_out  = out + 2 * cacheElems;

  unsigned char* ws = (unsigned char*)d_ws;
  constexpr size_t QB_BYTES = (size_t)Bc * Nc * Tc * Hc * 2;  // 16 MiB
  constexpr size_t KB_BYTES = (size_t)Bc * Kc * Tc * Hc * 2;  // 4 MiB
  u16* qb = (u16*)(ws);
  u16* kb = (u16*)(ws + QB_BYTES);
  u16* vb = (u16*)(ws + QB_BYTES + KB_BYTES);
  u16* ab = (u16*)(ws + QB_BYTES + 2 * KB_BYTES);

  hipMemcpyAsync(kc_out, kc_in, cacheElems * sizeof(float),
                 hipMemcpyDeviceToDevice, stream);
  hipMemcpyAsync(vc_out, vc_in, cacheElems * sizeof(float),
                 hipMemcpyDeviceToDevice, stream);

  qkv_rope_cache_kernel<<<dim3((Bc * Tc) / 64, Nc + 2 * Kc), 128, 0, stream>>>(
      x, positions, wq, wk, wv, widx, kc_out, vc_out, qb, kb, vb);

  flash_attn_kernel<<<dim3(Tc / 64, Bc * Nc), 128, 0, stream>>>(qb, kb, vb, ab);

  oproj_kernel<<<dim3((Bc * Tc) / 64, Dc / 128), 128, 0, stream>>>(ab, wo, o_out);
}
